// Attention3D_33303176413567
// MI455X (gfx1250) — matrix operand order repaired and checked by reading, not compiled, not run on hardware
//
#include <hip/hip_runtime.h>
#include <hip/hip_bf16.h>
#include <math.h>

// ---------------------------------------------------------------------------
// Attention3D for MI455X (gfx1250, wave32, WMMA 16x16x32 f16->f32)
// B=2, C=256, heads=4, hd=64, N=16^3=4096, groups=8
// ---------------------------------------------------------------------------

#define BB   2
#define CC   256
#define NH   4
#define HD   64
#define NG   8
#define CPG  (CC / NG)      // 32
#define NN   4096
#define EPSF 1e-5f

// padded LDS row strides (see analysis: avoid 64-bank conflicts, keep 16B align)
#define SRAW_STRIDE 33      // floats per row
#define PSH_STRIDE  48      // halfs per row (96 B, multiple of 16 B)

typedef _Float16 v8h  __attribute__((ext_vector_type(8)));
typedef _Float16 v16h __attribute__((ext_vector_type(16)));
typedef float    v8f  __attribute__((ext_vector_type(8)));

__device__ __forceinline__ v16h ld_v16h(const _Float16* p) {
    return *reinterpret_cast<const v16h*>(p);
}
__device__ __forceinline__ v8h ld_v8h(const _Float16* p) {
    return *reinterpret_cast<const v8h*>(p);
}
// A operands use contiguous k (16*hw + s, same order as B): combine two adjacent
// 8-half chunks into one v16h register group.
__device__ __forceinline__ v16h comb(v8h lo, v8h hi) {
    v16h r;
#pragma unroll
    for (int i = 0; i < 8; ++i) { r[i] = lo[i]; r[i + 8] = hi[i]; }
    return r;
}
__device__ __forceinline__ v8f wmma16(v16h a, v16h b, v8f c) {
    // D = A(16x32 f16) * B(32x16 f16) + C(16x16 f32)
    return __builtin_amdgcn_wmma_f32_16x16x32_f16(false, a, false, b,
                                                  (short)0, c, false, false);
}

// ---------------------------------------------------------------------------
// Kernel 1: per-group mean / rstd.  Group data is contiguous (channels-major,
// spatial innermost) => perfectly coalesced streaming reduction.
// ---------------------------------------------------------------------------
__global__ __launch_bounds__(256) void k_gn_stats(const float* __restrict__ x,
                                                  float* __restrict__ stats) {
    int grp = blockIdx.x;                 // b*NG + g, 16 total
    const float* xp = x + (size_t)grp * CPG * NN;
    float s = 0.f, ss = 0.f;
    for (int i = threadIdx.x; i < CPG * NN; i += 256) {
        float v = xp[i];
        s += v; ss += v * v;
    }
    __shared__ float sh_s[256], sh_q[256];
    sh_s[threadIdx.x] = s; sh_q[threadIdx.x] = ss;
    __syncthreads();
    for (int st = 128; st > 0; st >>= 1) {
        if (threadIdx.x < st) {
            sh_s[threadIdx.x] += sh_s[threadIdx.x + st];
            sh_q[threadIdx.x] += sh_q[threadIdx.x + st];
        }
        __syncthreads();
    }
    if (threadIdx.x == 0) {
        const float invn = 1.f / (float)(CPG * NN);
        float mean = sh_s[0] * invn;
        float var  = sh_q[0] * invn - mean * mean;
        stats[grp * 2 + 0] = mean;
        stats[grp * 2 + 1] = rsqrtf(var + EPSF);
    }
}

// ---------------------------------------------------------------------------
// Kernel 2: normalize + affine + transpose to [b][n][c] f16 (A-operand layout
// for the QKV GEMM).  64x64 LDS tile transpose keeps both sides coalesced.
// ---------------------------------------------------------------------------
__global__ __launch_bounds__(256) void k_norm_t(const float* __restrict__ x,
                                                const float* __restrict__ nw,
                                                const float* __restrict__ nb,
                                                const float* __restrict__ stats,
                                                _Float16* __restrict__ xnt) {
    __shared__ _Float16 tile[64][72];     // padded to avoid bank conflicts
    int b  = blockIdx.z;
    int c0 = blockIdx.y * 64;
    int n0 = blockIdx.x * 64;
    const float* xb = x + (size_t)b * CC * NN;
#pragma unroll
    for (int k = 0; k < 16; ++k) {
        int idx = k * 256 + threadIdx.x;
        int cc = idx >> 6, nn = idx & 63;
        int c = c0 + cc;
        int grp = b * NG + (c / CPG);
        float mean = stats[grp * 2 + 0];
        float rstd = stats[grp * 2 + 1];
        float v = xb[(size_t)c * NN + n0 + nn];
        v = (v - mean) * rstd * nw[c] + nb[c];
        tile[cc][nn] = (_Float16)v;
    }
    __syncthreads();
#pragma unroll
    for (int k = 0; k < 16; ++k) {
        int idx = k * 256 + threadIdx.x;
        int nn = idx >> 6, cc = idx & 63;
        xnt[((size_t)b * NN + n0 + nn) * CC + c0 + cc] = tile[cc][nn];
    }
}

// ---------------------------------------------------------------------------
// Kernel 3: f32 -> f16 weight conversion (qkv_w [768,256], proj_w [256,256]).
// ---------------------------------------------------------------------------
__global__ __launch_bounds__(256) void k_wconv(const float* __restrict__ qkv_w,
                                               const float* __restrict__ proj_w,
                                               _Float16* __restrict__ wq,
                                               _Float16* __restrict__ wp) {
    int i = blockIdx.x * 256 + threadIdx.x;
    if (i < 3 * CC * CC) wq[i] = (_Float16)qkv_w[i];
    if (i < CC * CC)     wp[i] = (_Float16)proj_w[i];
}

// ---------------------------------------------------------------------------
// Kernel 4: QKV GEMM.  y[n,o] = sum_c xnt[n,c] * w[o,c] + b[o].
// One wave per 16(n) x 64(o) tile: 4 accumulators share each A load
// (4x A-operand reuse), 32 chained WMMAs over K=256.
// The 64-wide o-tile is head-aligned (HD=64) so q/k/v routing is uniform.
// q,k stored [b][h][n][hd] (q pre-scaled by hd^-0.5); v stored [b][h][hd][n].
// ---------------------------------------------------------------------------
__global__ __launch_bounds__(32) void k_qkv(const _Float16* __restrict__ xnt,
                                            const _Float16* __restrict__ wq,
                                            const float* __restrict__ qkv_b,
                                            _Float16* __restrict__ qb,
                                            _Float16* __restrict__ kb,
                                            _Float16* __restrict__ vt) {
    int lane = threadIdx.x;
    int colL = lane & 15, hw = lane >> 4;
    int n0 = blockIdx.x * 16;
    int o0 = blockIdx.y * 64;           // 12 y-blocks over 768 outputs
    int b  = blockIdx.z;

    const _Float16* xrow = xnt + ((size_t)b * NN + n0 + colL) * CC;
    const _Float16* wrow = wq + (size_t)(o0 + colL) * CC;

    v8f acc[4] = {{}, {}, {}, {}};
#pragma unroll
    for (int kc = 0; kc < 8; ++kc) {
        v16h a = comb(ld_v8h(xrow + kc * 32 + 16 * hw),
                      ld_v8h(xrow + kc * 32 + 16 * hw + 8));
#pragma unroll
        for (int t = 0; t < 4; ++t) {
            v16h w = ld_v16h(wrow + (size_t)t * 16 * CC + kc * 32 + 16 * hw);
            acc[t] = wmma16(a, w, acc[t]);
        }
    }

    int n   = n0 + colL;
    int seg = o0 / CC;                  // 0=q, 1=k, 2=v (uniform per block)
    int h   = (o0 % CC) / HD;           // head (uniform per block)

#pragma unroll
    for (int t = 0; t < 4; ++t) {
        int orow0 = o0 + t * 16 + 8 * hw;       // first of this lane's 8 rows
        int cbase = (t * 16 + 8 * hw);          // channel-in-head base
        float4 bv0 = *reinterpret_cast<const float4*>(qkv_b + orow0);
        float4 bv1 = *reinterpret_cast<const float4*>(qkv_b + orow0 + 4);
        float bias[8] = {bv0.x, bv0.y, bv0.z, bv0.w, bv1.x, bv1.y, bv1.z, bv1.w};
        if (seg == 0) {                  // ---- Q (scaled) ----
            v8h outv;
#pragma unroll
            for (int r = 0; r < 8; ++r)
                outv[r] = (_Float16)((acc[t][r] + bias[r]) * 0.125f); // 64^-0.5
            *reinterpret_cast<v8h*>(
                qb + (((size_t)b * NH + h) * NN + n) * HD + cbase) = outv;
        } else if (seg == 1) {           // ---- K ----
            v8h outv;
#pragma unroll
            for (int r = 0; r < 8; ++r)
                outv[r] = (_Float16)(acc[t][r] + bias[r]);
            *reinterpret_cast<v8h*>(
                kb + (((size_t)b * NH + h) * NN + n) * HD + cbase) = outv;
        } else {                         // ---- V (transposed [hd][n]) ----
#pragma unroll
            for (int r = 0; r < 8; ++r)
                vt[(((size_t)b * NH + h) * HD + cbase + r) * NN + n] =
                    (_Float16)(acc[t][r] + bias[r]);
        }
    }
}

// ---------------------------------------------------------------------------
// Kernel 5: flash attention.  One wave per (b, h, 32-query tile).
// Per 32-key chunk: 8 WMMAs (scores), online softmax (all 32 lanes, one row
// each) via LDS, 8 WMMAs (out) -- 16 WMMAs per K/V chunk load, 2x the
// arithmetic intensity of a 16-query tile.  Single-wave workgroup =>
// per-wave in-order DS gives the D-layout -> A-layout transpose barrier-free.
// ---------------------------------------------------------------------------
__global__ __launch_bounds__(32) void k_attn(const _Float16* __restrict__ qb,
                                             const _Float16* __restrict__ kb,
                                             const _Float16* __restrict__ vt,
                                             _Float16* __restrict__ ao) {
    __shared__ float    s_raw[32 * SRAW_STRIDE];
    __shared__ _Float16 p_sh[32 * PSH_STRIDE];
    __shared__ float    alpha_sh[32];
    __shared__ float    inv_sh[32];

    int lane = threadIdx.x;
    int colL = lane & 15, hw = lane >> 4;
    int n0 = blockIdx.x * 32;
    int h  = blockIdx.y;
    int b  = blockIdx.z;

    const _Float16* qp = qb + (((size_t)b * NH + h) * NN) * HD;
    const _Float16* kp = kb + (((size_t)b * NH + h) * NN) * HD;
    const _Float16* vp = vt + (((size_t)b * NH + h) * HD) * NN;

    // Query A-operands for both 16-row tiles, resident for the whole pass.
    v16h aq[2][2];
#pragma unroll
    for (int t = 0; t < 2; ++t) {
        const _Float16* qrow = qp + (size_t)(n0 + t * 16 + colL) * HD;
        aq[t][0] = comb(ld_v8h(qrow + 16 * hw),      ld_v8h(qrow + 16 * hw + 8));
        aq[t][1] = comb(ld_v8h(qrow + 32 + 16 * hw), ld_v8h(qrow + 32 + 16 * hw + 8));
    }

    v8f acc[2][4] = {{{}, {}, {}, {}}, {{}, {}, {}, {}}};
    float m_run = -1e30f, l_run = 0.f;   // lane owns query row n0 + lane

    for (int mt = 0; mt < NN / 32; ++mt) {
        int m0 = mt * 32;

        // Prefetch next K/V chunk into cache (global_prefetch_b8).
        if (mt + 1 < NN / 32) {
            __builtin_prefetch(kp + (size_t)(m0 + 32 + colL) * HD, 0, 1);
            __builtin_prefetch(vp + (size_t)colL * NN + m0 + 32, 0, 1);
        }

        // ---- K B-operands (shared by both query tiles) ----
        const _Float16* kr0 = kp + (size_t)(m0 + colL) * HD + 16 * hw;
        const _Float16* kr1 = kr0 + 16 * HD;
        v16h bk00 = ld_v16h(kr0);
        v16h bk01 = ld_v16h(kr0 + 32);
        v16h bk10 = ld_v16h(kr1);
        v16h bk11 = ld_v16h(kr1 + 32);

        // ---- scores: S[32n x 32m], four 16x16 D tiles, 8 WMMAs ----
#pragma unroll
        for (int t = 0; t < 2; ++t) {
            v8f s0 = {};
            s0 = wmma16(aq[t][0], bk00, s0);
            s0 = wmma16(aq[t][1], bk01, s0);
            v8f s1 = {};
            s1 = wmma16(aq[t][0], bk10, s1);
            s1 = wmma16(aq[t][1], bk11, s1);
#pragma unroll
            for (int r = 0; r < 8; ++r) {
                int nr = t * 16 + r + 8 * hw;
                s_raw[nr * SRAW_STRIDE + colL]      = s0[r];
                s_raw[nr * SRAW_STRIDE + 16 + colL] = s1[r];
            }
        }

        // ---- online softmax: every lane owns one of the 32 query rows ----
        {
            float vals[32];
            float tmax = -1e30f;
#pragma unroll
            for (int j = 0; j < 32; ++j) {
                vals[j] = s_raw[lane * SRAW_STRIDE + j];   // odd stride: no conflicts
                tmax = fmaxf(tmax, vals[j]);
            }
            float Mn    = fmaxf(m_run, tmax);
            float alpha = __expf(m_run - Mn);
            float ssum  = 0.f;
#pragma unroll
            for (int j = 0; j < 32; ++j) {
                float p = __expf(vals[j] - Mn);
                ssum += p;
                p_sh[lane * PSH_STRIDE + j] = (_Float16)p;
            }
            l_run = l_run * alpha + ssum;
            m_run = Mn;
            alpha_sh[lane] = alpha;
        }

        // ---- rescale running output accumulators ----
#pragma unroll
        for (int t = 0; t < 2; ++t) {
            float al[8];
#pragma unroll
            for (int r = 0; r < 8; ++r) al[r] = alpha_sh[t * 16 + r + 8 * hw];
#pragma unroll
            for (int ct = 0; ct < 4; ++ct)
#pragma unroll
                for (int r = 0; r < 8; ++r) acc[t][ct][r] *= al[r];
        }

        // ---- out += P(16x32) x V(32m x 16c): 2 tiles x 4 c-tiles, 8 WMMAs ----
        const _Float16* vrow = vp + (size_t)colL * NN + m0 + 16 * hw;
        v16h bv0 = ld_v16h(vrow);
        v16h bv1 = ld_v16h(vrow + 16 * NN);
        v16h bv2 = ld_v16h(vrow + 32 * NN);
        v16h bv3 = ld_v16h(vrow + 48 * NN);
#pragma unroll
        for (int t = 0; t < 2; ++t) {
            const _Float16* prow = p_sh + (size_t)(t * 16 + colL) * PSH_STRIDE;
            v16h ap = comb(ld_v8h(prow + 16 * hw), ld_v8h(prow + 16 * hw + 8));
            acc[t][0] = wmma16(ap, bv0, acc[t][0]);
            acc[t][1] = wmma16(ap, bv1, acc[t][1]);
            acc[t][2] = wmma16(ap, bv2, acc[t][2]);
            acc[t][3] = wmma16(ap, bv3, acc[t][3]);
        }
    }

    inv_sh[lane] = 1.f / l_run;

    // ao layout [b][n][C] f16 (A-operand layout for the projection GEMM)
    _Float16* aob = ao + (size_t)b * NN * CC;
#pragma unroll
    for (int t = 0; t < 2; ++t) {
        float inv[8];
#pragma unroll
        for (int r = 0; r < 8; ++r) inv[r] = inv_sh[t * 16 + r + 8 * hw];
#pragma unroll
        for (int r = 0; r < 8; ++r) {
            int n = n0 + t * 16 + r + 8 * hw;
            _Float16* dst = aob + (size_t)n * CC + h * HD + colL;
            dst[0]  = (_Float16)(acc[t][0][r] * inv[r]);
            dst[16] = (_Float16)(acc[t][1][r] * inv[r]);
            dst[32] = (_Float16)(acc[t][2][r] * inv[r]);
            dst[48] = (_Float16)(acc[t][3][r] * inv[r]);
        }
    }
}

// ---------------------------------------------------------------------------
// Kernel 6: projection + bias + residual.  y[n,o] = sum_c ao[n,c]*wp[o,c].
// One wave per 16(n) x 64(o) tile (4x A reuse, 32 WMMAs).
// Output written in reference layout [b][C][N] f32, float4-vectorized.
// ---------------------------------------------------------------------------
__global__ __launch_bounds__(32) void k_proj(const _Float16* __restrict__ ao,
                                             const _Float16* __restrict__ wp,
                                             const float* __restrict__ proj_b,
                                             const float* __restrict__ x,
                                             float* __restrict__ out) {
    int lane = threadIdx.x;
    int colL = lane & 15, hw = lane >> 4;
    int n0 = blockIdx.x * 16;
    int o0 = blockIdx.y * 64;
    int b  = blockIdx.z;

    const _Float16* arow = ao + ((size_t)b * NN + n0 + colL) * CC;
    const _Float16* wrow = wp + (size_t)(o0 + colL) * CC;

    v8f acc[4] = {{}, {}, {}, {}};
#pragma unroll
    for (int kc = 0; kc < 8; ++kc) {
        v16h a = comb(ld_v8h(arow + kc * 32 + 16 * hw),
                      ld_v8h(arow + kc * 32 + 16 * hw + 8));
#pragma unroll
        for (int t = 0; t < 4; ++t) {
            v16h w = ld_v16h(wrow + (size_t)t * 16 * CC + kc * 32 + 16 * hw);
            acc[t] = wmma16(a, w, acc[t]);
        }
    }

    int nbase = n0 + 8 * hw;
#pragma unroll
    for (int t = 0; t < 4; ++t) {
        int o = o0 + t * 16 + colL;
        float bias = proj_b[o];
        const float* xr = x + ((size_t)b * CC + o) * NN + nbase;
        float4 x0 = *reinterpret_cast<const float4*>(xr);
        float4 x1 = *reinterpret_cast<const float4*>(xr + 4);
        float4 r0 = {acc[t][0] + bias + x0.x, acc[t][1] + bias + x0.y,
                     acc[t][2] + bias + x0.z, acc[t][3] + bias + x0.w};
        float4 r1 = {acc[t][4] + bias + x1.x, acc[t][5] + bias + x1.y,
                     acc[t][6] + bias + x1.z, acc[t][7] + bias + x1.w};
        float* op = out + ((size_t)b * CC + o) * NN + nbase;
        *reinterpret_cast<float4*>(op)     = r0;
        *reinterpret_cast<float4*>(op + 4) = r1;
    }
}

// ---------------------------------------------------------------------------
// Workspace layout (bytes)
// ---------------------------------------------------------------------------
static constexpr size_t OFF_STATS = 0;                              // 128 B
static constexpr size_t OFF_XNT   = 256;
static constexpr size_t SZ_XNT    = (size_t)BB * NN * CC * 2;       // 4 MiB
static constexpr size_t OFF_WQ    = OFF_XNT + SZ_XNT;
static constexpr size_t SZ_WQ     = (size_t)3 * CC * CC * 2;        // 384 KiB
static constexpr size_t OFF_WP    = OFF_WQ + SZ_WQ;
static constexpr size_t SZ_WP     = (size_t)CC * CC * 2;            // 128 KiB
static constexpr size_t OFF_Q     = OFF_WP + SZ_WP;
static constexpr size_t SZ_QKV    = (size_t)BB * NH * NN * HD * 2;  // 4 MiB
static constexpr size_t OFF_K     = OFF_Q + SZ_QKV;
static constexpr size_t OFF_VT    = OFF_K + SZ_QKV;
static constexpr size_t OFF_AO    = OFF_VT + SZ_QKV;

extern "C" void kernel_launch(void* const* d_in, const int* in_sizes, int n_in,
                              void* d_out, int out_size, void* d_ws, size_t ws_size,
                              hipStream_t stream) {
    (void)in_sizes; (void)n_in; (void)out_size; (void)ws_size;
    const float* x      = (const float*)d_in[0];
    const float* norm_w = (const float*)d_in[1];
    const float* norm_b = (const float*)d_in[2];
    const float* qkv_w  = (const float*)d_in[3];
    const float* qkv_b  = (const float*)d_in[4];
    const float* proj_w = (const float*)d_in[5];
    const float* proj_b = (const float*)d_in[6];
    float* out = (float*)d_out;

    char* ws = (char*)d_ws;
    float*    stats = (float*)(ws + OFF_STATS);
    _Float16* xnt   = (_Float16*)(ws + OFF_XNT);
    _Float16* wq    = (_Float16*)(ws + OFF_WQ);
    _Float16* wp    = (_Float16*)(ws + OFF_WP);
    _Float16* qbuf  = (_Float16*)(ws + OFF_Q);
    _Float16* kbuf  = (_Float16*)(ws + OFF_K);
    _Float16* vtb   = (_Float16*)(ws + OFF_VT);
    _Float16* aob   = (_Float16*)(ws + OFF_AO);

    k_gn_stats<<<BB * NG, 256, 0, stream>>>(x, stats);
    k_wconv<<<(3 * CC * CC + 255) / 256, 256, 0, stream>>>(qkv_w, proj_w, wq, wp);
    k_norm_t<<<dim3(NN / 64, CC / 64, BB), 256, 0, stream>>>(x, norm_w, norm_b, stats, xnt);
    k_qkv<<<dim3(NN / 16, 12, BB), 32, 0, stream>>>(xnt, wq, qkv_b, qbuf, kbuf, vtb);
    k_attn<<<dim3(NN / 32, NH, BB), 32, 0, stream>>>(qbuf, kbuf, vtb, aob);
    k_proj<<<dim3(NN / 16, 4, BB), 32, 0, stream>>>(aob, wp, proj_b, x, out);
}
